// NodeModel_42047729828006
// MI455X (gfx1250) — matrix operand order repaired and checked by reading, not compiled, not run on hardware
//
#include <hip/hip_runtime.h>

// ---------------------------------------------------------------- types
typedef __bf16 bf16_t;
typedef __attribute__((ext_vector_type(16))) __bf16 v16bf;
typedef __attribute__((ext_vector_type(8)))  __bf16 v8bf;
typedef __attribute__((ext_vector_type(8)))  float  v8f;

#define N_NODES  50000
#define N_EDGES  800000
#define NODE_DIM 64
#define EDGE_DIM 64
#define HIDDEN   128
#define FLOW_DIM 64

#define TILE 64          // rows (edges or nodes) per block
#define LDA  136         // LDS row stride in bf16 elems (272B = 17*16B, keeps 16B align)

// ---------------------------------------------------------------- helpers
__device__ __forceinline__ v8f wmma_bf16(v16bf a, v16bf b, v8f c) {
  return __builtin_amdgcn_wmma_f32_16x16x32_bf16(
      /*neg_a=*/false, a, /*neg_b=*/false, b,
      /*c_mod=*/(short)0, c, /*reuse_a=*/false, /*reuse_b=*/false);
}

// A-matrix (16x32 bf16) fragment from row-major LDS tile.
// lane<16: M=lane,    K = kt*32 + {0..7, 16..23}
// lane>=16: M=lane-16, K = kt*32 + {8..15, 24..31}
__device__ __forceinline__ v16bf load_a_frag(const bf16_t* lds, int rowBase,
                                             int kt, int lane) {
  int m  = lane & 15;
  int c0 = kt * 32 + ((lane >> 4) << 3);
  const bf16_t* p = lds + (rowBase + m) * LDA + c0;
  v8bf lo = *(const v8bf*)(p);
  v8bf hi = *(const v8bf*)(p + 16);
  v16bf a;
#pragma unroll
  for (int j = 0; j < 8; ++j) { a[j] = lo[j]; a[j + 8] = hi[j]; }
  return a;
}

// B-matrix fragment from pre-packed global weights: 32B contiguous per lane.
__device__ __forceinline__ v16bf load_b_frag(const bf16_t* __restrict__ packed,
                                             int frag, int lane) {
  const bf16_t* p = packed + ((size_t)frag * 32 + lane) * 16;
  v8bf lo = *(const v8bf*)(p);
  v8bf hi = *(const v8bf*)(p + 8);
  v16bf b;
#pragma unroll
  for (int j = 0; j < 8; ++j) { b[j] = lo[j]; b[j + 8] = hi[j]; }
  return b;
}

// ---------------------------------------------------------------- kernels
__global__ void zero_flow_kernel(float4* __restrict__ p) {
  size_t i = (size_t)blockIdx.x * blockDim.x + threadIdx.x;   // 1.6M float4
  p[i] = make_float4(0.f, 0.f, 0.f, 0.f);
}

// Re-pack a [K,N] row-major f32 weight matrix into bf16 WMMA B-fragment layout.
__global__ void pack_weight_kernel(const float* __restrict__ W,
                                   bf16_t* __restrict__ dst, int K, int N) {
  int idx = blockIdx.x * blockDim.x + threadIdx.x;
  if (idx >= K * N) return;
  int frag = idx >> 9;           // 512 elems per fragment (32 lanes * 16)
  int rem  = idx & 511;
  int lane = rem >> 4;
  int j    = rem & 15;
  int ntiles = N >> 4;
  int kt = frag / ntiles, nt = frag % ntiles;
  int n  = nt * 16 + (lane & 15);
  int k  = kt * 32 + ((lane >> 4) << 3) + (j & 7) + ((j >> 3) << 4);
  dst[idx] = (bf16_t)W[(size_t)k * N + n];
}

// ---- edge kernel: two MLPs over gathered [x[col] || edge_attr], scatter-add.
__global__ __launch_bounds__(128)
void edge_flow_kernel(const float* __restrict__ x,
                      const long long* __restrict__ edge_index,
                      const float* __restrict__ edge_attr,
                      const bf16_t* __restrict__ Wo1p, const float* __restrict__ bo1,
                      const bf16_t* __restrict__ Wo2p, const float* __restrict__ bo2,
                      const bf16_t* __restrict__ Wi1p, const float* __restrict__ bi1,
                      const bf16_t* __restrict__ Wi2p, const float* __restrict__ bi2,
                      float* __restrict__ flow) {
  __shared__ bf16_t Ain[TILE * LDA];
  __shared__ bf16_t Hs [TILE * LDA];
  __shared__ int rowIds[TILE], colIds[TILE];

  const int tid   = threadIdx.x;
  const int lane  = tid & 31;
  const int wave  = tid >> 5;          // 0..3
  const int eBase = blockIdx.x * TILE;

  if (tid < TILE) {
    rowIds[tid] = (int)edge_index[eBase + tid];
    colIds[tid] = (int)edge_index[(size_t)N_EDGES + eBase + tid];
  }
  __syncthreads();

  // Stage edge_in = [x[col] (64) || edge_attr (64)] as bf16 into LDS.
  for (int i = tid; i < TILE * 32; i += 128) {
    int r = i >> 5;
    int c = (i & 31) * 4;
    const float* src = (c < 64)
        ? (x + (size_t)colIds[r] * NODE_DIM + c)
        : (edge_attr + (size_t)(eBase + r) * EDGE_DIM + (c - 64));
    float4 v = *(const float4*)src;
    bf16_t* dst = &Ain[r * LDA + c];
    dst[0] = (bf16_t)v.x; dst[1] = (bf16_t)v.y;
    dst[2] = (bf16_t)v.z; dst[3] = (bf16_t)v.w;
  }
  __syncthreads();

  const int rowBase = wave * 16;
  v16bf afr[4];
#pragma unroll
  for (int kt = 0; kt < 4; ++kt) afr[kt] = load_a_frag(Ain, rowBase, kt, lane);

#pragma unroll
  for (int path = 0; path < 2; ++path) {
    const bf16_t* W1p = path == 0 ? Wo1p : Wi1p;
    const float*  b1  = path == 0 ? bo1  : bi1;
    const bf16_t* W2p = path == 0 ? Wo2p : Wi2p;
    const float*  b2  = path == 0 ? bo2  : bi2;
    const int colOff  = path == 0 ? 64 : 0;   // concat(flow_in, flow_out)

    // Layer 1: [16,128] x [128,128] -> relu -> LDS (bf16)
#pragma unroll
    for (int nt = 0; nt < 8; ++nt) {
      v8f acc = {};
#pragma unroll
      for (int kt = 0; kt < 4; ++kt)
        acc = wmma_bf16(afr[kt], load_b_frag(W1p, kt * 8 + nt, lane), acc);
      float bias = b1[nt * 16 + (lane & 15)];
#pragma unroll
      for (int r = 0; r < 8; ++r) {
        int m = r + ((lane >> 4) << 3);
        float v = acc[r] + bias;
        Hs[(rowBase + m) * LDA + nt * 16 + (lane & 15)] = (bf16_t)(v > 0.f ? v : 0.f);
      }
    }

    // Layer 2: [16,128] x [128,64] -> relu -> masked atomic scatter
#pragma unroll
    for (int nt = 0; nt < 4; ++nt) {
      v8f acc = {};
#pragma unroll
      for (int kt = 0; kt < 4; ++kt)
        acc = wmma_bf16(load_a_frag(Hs, rowBase, kt, lane),
                        load_b_frag(W2p, kt * 4 + nt, lane), acc);
      float bias = b2[nt * 16 + (lane & 15)];
#pragma unroll
      for (int r = 0; r < 8; ++r) {
        int m   = r + ((lane >> 4) << 3);
        int rid = rowIds[rowBase + m];
        int cid = colIds[rowBase + m];
        bool keep = (path == 0) ? (rid < cid) : (rid > cid);
        float v = acc[r] + bias;
        v = v > 0.f ? v : 0.f;
        if (keep && v != 0.f)
          atomicAdd(&flow[(size_t)rid * 128 + colOff + nt * 16 + (lane & 15)], v);
      }
    }
  }
}

// ---- node kernel: out = mlp2(flow; Wn)
__global__ __launch_bounds__(128)
void node_mlp_kernel(const float* __restrict__ flow,
                     const bf16_t* __restrict__ Wn1p, const float* __restrict__ bn1,
                     const bf16_t* __restrict__ Wn2p, const float* __restrict__ bn2,
                     float* __restrict__ out) {
  __shared__ bf16_t Ain[TILE * LDA];
  __shared__ bf16_t Hs [TILE * LDA];

  const int tid   = threadIdx.x;
  const int lane  = tid & 31;
  const int wave  = tid >> 5;
  const int nBase = blockIdx.x * TILE;

  for (int i = tid; i < TILE * 32; i += 128) {
    int r = i >> 5;
    int c = (i & 31) * 4;
    int node = nBase + r;
    float4 v = make_float4(0.f, 0.f, 0.f, 0.f);
    if (node < N_NODES) v = *(const float4*)(flow + (size_t)node * 128 + c);
    bf16_t* dst = &Ain[r * LDA + c];
    dst[0] = (bf16_t)v.x; dst[1] = (bf16_t)v.y;
    dst[2] = (bf16_t)v.z; dst[3] = (bf16_t)v.w;
  }
  __syncthreads();

  const int rowBase = wave * 16;
  v16bf afr[4];
#pragma unroll
  for (int kt = 0; kt < 4; ++kt) afr[kt] = load_a_frag(Ain, rowBase, kt, lane);

  // Layer 1
#pragma unroll
  for (int nt = 0; nt < 8; ++nt) {
    v8f acc = {};
#pragma unroll
    for (int kt = 0; kt < 4; ++kt)
      acc = wmma_bf16(afr[kt], load_b_frag(Wn1p, kt * 8 + nt, lane), acc);
    float bias = bn1[nt * 16 + (lane & 15)];
#pragma unroll
    for (int r = 0; r < 8; ++r) {
      int m = r + ((lane >> 4) << 3);
      float v = acc[r] + bias;
      Hs[(rowBase + m) * LDA + nt * 16 + (lane & 15)] = (bf16_t)(v > 0.f ? v : 0.f);
    }
  }

  // Layer 2
#pragma unroll
  for (int nt = 0; nt < 4; ++nt) {
    v8f acc = {};
#pragma unroll
    for (int kt = 0; kt < 4; ++kt)
      acc = wmma_bf16(load_a_frag(Hs, rowBase, kt, lane),
                      load_b_frag(Wn2p, kt * 4 + nt, lane), acc);
    float bias = bn2[nt * 16 + (lane & 15)];
#pragma unroll
    for (int r = 0; r < 8; ++r) {
      int m = r + ((lane >> 4) << 3);
      int node = nBase + rowBase + m;
      float v = acc[r] + bias;
      if (node < N_NODES)
        out[(size_t)node * NODE_DIM + nt * 16 + (lane & 15)] = (v > 0.f ? v : 0.f);
    }
  }
}

// ---------------------------------------------------------------- launch
extern "C" void kernel_launch(void* const* d_in, const int* in_sizes, int n_in,
                              void* d_out, int out_size, void* d_ws, size_t ws_size,
                              hipStream_t stream) {
  const float*     x          = (const float*)d_in[0];
  const long long* edge_index = (const long long*)d_in[1];
  const float*     edge_attr  = (const float*)d_in[2];
  const float* Wo1 = (const float*)d_in[3];  const float* bo1 = (const float*)d_in[4];
  const float* Wo2 = (const float*)d_in[5];  const float* bo2 = (const float*)d_in[6];
  const float* Wi1 = (const float*)d_in[7];  const float* bi1 = (const float*)d_in[8];
  const float* Wi2 = (const float*)d_in[9];  const float* bi2 = (const float*)d_in[10];
  const float* Wn1 = (const float*)d_in[11]; const float* bn1 = (const float*)d_in[12];
  const float* Wn2 = (const float*)d_in[13]; const float* bn2 = (const float*)d_in[14];
  float* out = (float*)d_out;

  // workspace: flow accumulator [N,128] f32, then packed bf16 weights
  float*  flow  = (float*)d_ws;
  bf16_t* wbase = (bf16_t*)((char*)d_ws + (size_t)N_NODES * 128 * sizeof(float));
  bf16_t* Wo1p = wbase;              // 128x128 -> 16384
  bf16_t* Wo2p = Wo1p + 16384;       // 128x64  -> 8192
  bf16_t* Wi1p = Wo2p + 8192;
  bf16_t* Wi2p = Wi1p + 16384;
  bf16_t* Wn1p = Wi2p + 8192;
  bf16_t* Wn2p = Wn1p + 16384;

  // 1) zero flow accumulator (1.6M float4 exactly)
  zero_flow_kernel<<<(N_NODES * 128 / 4) / 256, 256, 0, stream>>>((float4*)flow);

  // 2) pack weights to bf16 B-fragment layout
  pack_weight_kernel<<<(128 * 128 + 255) / 256, 256, 0, stream>>>(Wo1, Wo1p, 128, 128);
  pack_weight_kernel<<<(128 *  64 + 255) / 256, 256, 0, stream>>>(Wo2, Wo2p, 128, 64);
  pack_weight_kernel<<<(128 * 128 + 255) / 256, 256, 0, stream>>>(Wi1, Wi1p, 128, 128);
  pack_weight_kernel<<<(128 *  64 + 255) / 256, 256, 0, stream>>>(Wi2, Wi2p, 128, 64);
  pack_weight_kernel<<<(128 * 128 + 255) / 256, 256, 0, stream>>>(Wn1, Wn1p, 128, 128);
  pack_weight_kernel<<<(128 *  64 + 255) / 256, 256, 0, stream>>>(Wn2, Wn2p, 128, 64);

  // 3) edge MLPs + masked scatter-add (800000 / 64 = 12500 blocks)
  edge_flow_kernel<<<N_EDGES / TILE, 128, 0, stream>>>(
      x, edge_index, edge_attr,
      Wo1p, bo1, Wo2p, bo2, Wi1p, bi1, Wi2p, bi2, flow);

  // 4) node MLP (ceil(50000/64) = 782 blocks)
  node_mlp_kernel<<<(N_NODES + TILE - 1) / TILE, 128, 0, stream>>>(
      flow, Wn1p, bn1, Wn2p, bn2, out);
}
